// TaggingPartialCRFLossBuilder_67121748902288
// MI455X (gfx1250) — matrix operand order repaired and checked by reading, not compiled, not run on hardware
//
#include <hip/hip_runtime.h>
#include <stdint.h>

#define Bn 256
#define Tn 512
#define Kn 128
#define NEGB  (-1.0e30f)
#define LOG2E 1.4426950408889634f
#define LN2   0.6931471805599453f

typedef __attribute__((ext_vector_type(16))) __bf16 v16bf;
typedef __attribute__((ext_vector_type(8)))  float  v8f;

static __device__ __forceinline__ float max4(float4 a) {
    return fmaxf(fmaxf(a.x, a.y), fmaxf(a.z, a.w));
}

// One workgroup = 256 threads = 8 waves, owns 16 batch rows of one pass
// (pass 0 = raw partition, pass 1 = gold-clamped partition).
// Recursion runs in log2 domain (raw v_exp_f32 / v_log_f32):
//   a2'[j] = m2 + log2( exp2(a2-m2) @ exp2(trans*log2e) )[j] + emit*log2e
// GEMM on v_wmma_f32_16x16x32_bf16; emissions stream in TRANSPOSED via
// per-lane global_load_async_to_lds_b32 (ASYNCcnt) hidden behind phases 1-2.
__global__ __launch_bounds__(256)
void crf_scan_kernel(const float* __restrict__ tw,
                     const float* __restrict__ trans,
                     const int*   __restrict__ gold,
                     const int*   __restrict__ lengths,
                     float*       __restrict__ ws)
{
    __shared__ __align__(32) __bf16 EtS[Kn * Kn];   // Et[j][i] = exp(trans[i][j]) (N-major B operand)
    __shared__ __align__(32) __bf16 Ubf[16 * Kn];   // u = exp2(a2 - m2), A operand
    __shared__ __align__(16) float  alphaS[16 * Kn];// fp32 running alpha (log2 domain), row-major
    __shared__ __align__(16) float  emitT[Kn * 16]; // emission tile, TRANSPOSED: emitT[j][r]
    __shared__ __align__(16) float  part[256];
    __shared__ __align__(16) float  mrow[16];
    __shared__ __align__(16) int    goldS[16];

    const int tid  = threadIdx.x;
    const int lane = tid & 31;
    const int wv   = tid >> 5;          // wave 0..7 -> output column tile
    const int wg   = blockIdx.x;        // 0..31
    const int pass = wg >> 4;           // 0 = raw, 1 = clamped
    const int b0   = (wg & 15) * 16;    // batch tile base

    // ---- one-time: E^T = exp(trans)^T into LDS (bf16) ----
    for (int idx = tid; idx < Kn * Kn; idx += 256) {
        int i = idx >> 7;
        int j = idx & (Kn - 1);
        EtS[j * Kn + i] = (__bf16)__builtin_amdgcn_exp2f(trans[idx] * LOG2E);
    }

    // ---- alpha0 = tag_weights[:,0] * log2e (+ clamp for pass 1) ----
    {
        int r = tid >> 4, q = tid & 15;
        int b = b0 + r;
        int g0 = gold[b * Tn + 0];
        for (int k = 0; k < 8; ++k) {
            int c = q * 8 + k;
            float v = tw[((size_t)b * Tn + 0) * Kn + c] * LOG2E;
            bool bad = (pass != 0) & (g0 >= 0) & (g0 != c);
            alphaS[r * Kn + c] = bad ? NEGB : v;
        }
    }

    // ---- async-copy descriptors: thread owns row r = tid>>4, cols jc+16u ----
    // offset: immediate adds to BOTH global and LDS addresses, so pre-subtract
    // 64*u from the LDS base and share one 64-bit global base register.
    const int rr = tid >> 4, jc = tid & 15;
    const uint32_t emit_lds = (uint32_t)(uintptr_t)&emitT[0];
    uint32_t lds8[8];
    #pragma unroll
    for (int u = 0; u < 8; ++u)
        lds8[u] = emit_lds + (uint32_t)(((jc + 16 * u) * 16 + rr) * 4) - 64u * u;
    const uint64_t gbase = (uint64_t)(uintptr_t)(tw + ((size_t)(b0 + rr) * Tn) * Kn + jc);

    const int l15  = lane & 15;              // A row / B+C column
    const int moff = (lane < 16) ? 0 : 8;    // C/D row group
    const int j0   = wv * 16;
    int lenr[8];
    #pragma unroll
    for (int v = 0; v < 8; ++v) lenr[v] = lengths[b0 + v + moff];

    __syncthreads();

    // lane-private copy of the 8 alpha values this lane owns (M=moff+v, j=j0+l15)
    float aprev[8];
    #pragma unroll
    for (int v = 0; v < 8; ++v) aprev[v] = alphaS[(v + moff) * Kn + (j0 + l15)];

    for (int t = 1; t < Tn; ++t) {
        // phase 0: async transposed emission-tile copy for this step (ASYNCcnt)
        {
            uint64_t g = gbase + (uint64_t)t * (Kn * sizeof(float));
            asm volatile("global_load_async_to_lds_b32 %0, %1, off"            :: "v"(lds8[0]), "v"(g) : "memory");
            asm volatile("global_load_async_to_lds_b32 %0, %1, off offset:64"  :: "v"(lds8[1]), "v"(g) : "memory");
            asm volatile("global_load_async_to_lds_b32 %0, %1, off offset:128" :: "v"(lds8[2]), "v"(g) : "memory");
            asm volatile("global_load_async_to_lds_b32 %0, %1, off offset:192" :: "v"(lds8[3]), "v"(g) : "memory");
            asm volatile("global_load_async_to_lds_b32 %0, %1, off offset:256" :: "v"(lds8[4]), "v"(g) : "memory");
            asm volatile("global_load_async_to_lds_b32 %0, %1, off offset:320" :: "v"(lds8[5]), "v"(g) : "memory");
            asm volatile("global_load_async_to_lds_b32 %0, %1, off offset:384" :: "v"(lds8[6]), "v"(g) : "memory");
            asm volatile("global_load_async_to_lds_b32 %0, %1, off offset:448" :: "v"(lds8[7]), "v"(g) : "memory");
        }
        if (tid < 16) goldS[tid] = gold[(b0 + tid) * Tn + t];

        // phase 1: partial row maxima of alpha (2x ds_load_b128)
        {
            float4 a0 = *(const float4*)&alphaS[rr * Kn + jc * 8];
            float4 a1 = *(const float4*)&alphaS[rr * Kn + jc * 8 + 4];
            part[tid] = fmaxf(max4(a0), max4(a1));
        }
        __syncthreads();

        // phase 2: full row max m2 (4x b128), build U = bf16(exp2(a2 - m2))
        {
            float4 p0 = *(const float4*)&part[rr * 16 + 0];
            float4 p1 = *(const float4*)&part[rr * 16 + 4];
            float4 p2 = *(const float4*)&part[rr * 16 + 8];
            float4 p3 = *(const float4*)&part[rr * 16 + 12];
            float m = fmaxf(fmaxf(max4(p0), max4(p1)), fmaxf(max4(p2), max4(p3)));
            if (jc == 0) mrow[rr] = m;
            float4 a0 = *(const float4*)&alphaS[rr * Kn + jc * 8];
            float4 a1 = *(const float4*)&alphaS[rr * Kn + jc * 8 + 4];
            float av[8] = {a0.x, a0.y, a0.z, a0.w, a1.x, a1.y, a1.z, a1.w};
            #pragma unroll
            for (int k = 0; k < 8; ++k)
                Ubf[rr * Kn + jc * 8 + k] = (__bf16)__builtin_amdgcn_exp2f(av[k] - m);
        }
        asm volatile("s_wait_asynccnt 0x0" ::: "memory");  // our emit chunks landed
        __syncthreads();                                   // everyone's landed + U ready

        // phase 3: S = U @ E on the matrix pipe (K=128 -> 4 accumulating WMMAs)
        v8f c = {};
        #pragma unroll
        for (int kk = 0; kk < 4; ++kk) {
            v16bf a, bm;
            int abase = l15 * Kn + kk * 32 + ((lane < 16) ? 0 : 16);
            #pragma unroll
            for (int e = 0; e < 8; ++e) {
                a[e]     = Ubf[abase + e];
                a[e + 8] = Ubf[abase + 8 + e];
            }
            bm = *(const v16bf*)&EtS[(j0 + l15) * Kn + kk * 32 + ((lane < 16) ? 0 : 16)];
            c = __builtin_amdgcn_wmma_f32_16x16x32_bf16(
                    false, a, false, bm, (short)0, c, false, false);
        }

        // contiguous b128 LDS reads for this lane: emissions, gold, row maxima
        const int jcol = j0 + l15;
        float4 e0 = *(const float4*)&emitT[jcol * 16 + moff];
        float4 e1 = *(const float4*)&emitT[jcol * 16 + moff + 4];
        int4   g0 = *(const int4*)&goldS[moff];
        int4   g1 = *(const int4*)&goldS[moff + 4];
        float4 m0 = *(const float4*)&mrow[moff];
        float4 m1 = *(const float4*)&mrow[moff + 4];
        float emitv[8] = {e0.x, e0.y, e0.z, e0.w, e1.x, e1.y, e1.z, e1.w};
        int   goldv[8] = {g0.x, g0.y, g0.z, g0.w, g1.x, g1.y, g1.z, g1.w};
        float mrowv[8] = {m0.x, m0.y, m0.z, m0.w, m1.x, m1.y, m1.z, m1.w};

        // alpha update: a2' = m2 + log2(S) + emit*log2e, length mask as pure cndmask
        #pragma unroll
        for (int v = 0; v < 8; ++v) {
            float emit = emitv[v] * LOG2E;
            bool bad = (pass != 0) & (goldv[v] >= 0) & (goldv[v] != jcol);
            emit = bad ? NEGB : emit;
            float nv = mrowv[v] + __builtin_amdgcn_logf(c[v]) + emit;
            asm volatile("" : "+v"(nv));   // opaque: forbid sinking under a branch
            aprev[v] = (t < lenr[v]) ? nv : aprev[v];
            alphaS[(v + moff) * Kn + jcol] = aprev[v];
        }
        __syncthreads();   // alpha + emitT safe for next iteration
    }

    // ---- final per-row logsumexp (scale back by ln2) -> ws[pass*Bn + b] ----
    {
        float4 a0 = *(const float4*)&alphaS[rr * Kn + jc * 8];
        float4 a1 = *(const float4*)&alphaS[rr * Kn + jc * 8 + 4];
        part[tid] = fmaxf(max4(a0), max4(a1));
        __syncthreads();
        float4 p0 = *(const float4*)&part[rr * 16 + 0];
        float4 p1 = *(const float4*)&part[rr * 16 + 4];
        float4 p2 = *(const float4*)&part[rr * 16 + 8];
        float4 p3 = *(const float4*)&part[rr * 16 + 12];
        float m = fmaxf(fmaxf(max4(p0), max4(p1)), fmaxf(max4(p2), max4(p3)));
        __syncthreads();
        float av[8] = {a0.x, a0.y, a0.z, a0.w, a1.x, a1.y, a1.z, a1.w};
        float ps = 0.f;
        #pragma unroll
        for (int k = 0; k < 8; ++k) ps += __builtin_amdgcn_exp2f(av[k] - m);
        part[tid] = ps;
        __syncthreads();
        if (jc == 0) {
            float s = 0.f;
            #pragma unroll
            for (int k = 0; k < 16; ++k) s += part[rr * 16 + k];
            ws[pass * Bn + b0 + rr] = (m + __builtin_amdgcn_logf(s)) * LN2;
        }
    }
}

__global__ __launch_bounds__(256)
void crf_finalize_kernel(const float* __restrict__ ws, float* __restrict__ out)
{
    __shared__ float red[256];
    int tid = threadIdx.x;
    red[tid] = ws[tid] - ws[Bn + tid];   // lse_raw - lse_clamped
    __syncthreads();
    for (int s = 128; s > 0; s >>= 1) {
        if (tid < s) red[tid] += red[tid + s];
        __syncthreads();
    }
    if (tid == 0) out[0] = red[0] * (1.0f / Bn);
}

extern "C" void kernel_launch(void* const* d_in, const int* in_sizes, int n_in,
                              void* d_out, int out_size, void* d_ws, size_t ws_size,
                              hipStream_t stream) {
    const float* tw      = (const float*)d_in[0];   // (B,T,K) f32
    const float* trans   = (const float*)d_in[1];   // (K,K)   f32
    const int*   gold    = (const int*)d_in[2];     // (B,T)   i32
    const int*   lengths = (const int*)d_in[3];     // (B,)    i32
    float*       ws      = (float*)d_ws;            // 2*Bn floats

    crf_scan_kernel<<<32, 256, 0, stream>>>(tw, trans, gold, lengths, ws);
    crf_finalize_kernel<<<1, 256, 0, stream>>>(ws, (float*)d_out);
}
